// HyperbolicPrototypicalLoss_47502338294337
// MI455X (gfx1250) — matrix operand order repaired and checked by reading, not compiled, not run on hardware
//
#include <hip/hip_runtime.h>
#include <hip/hip_bf16.h>

typedef __attribute__((ext_vector_type(16))) __bf16 v16bf;
typedef __attribute__((ext_vector_type(8)))  __bf16 v8bf;
typedef __attribute__((ext_vector_type(8)))  float  v8f;

#define BDIM 8192
#define KDIM 4096
#define DDIM 512
#define DP1  513

// ---------------------------------------------------------------------------
// Phase 0: split f32 space-components of Q and P into bf16 (hi, lo) pairs.
// ---------------------------------------------------------------------------
__global__ void split_bf16_kernel(const float* __restrict__ q,
                                  const float* __restrict__ p,
                                  __bf16* __restrict__ qhi, __bf16* __restrict__ qlo,
                                  __bf16* __restrict__ phi, __bf16* __restrict__ plo) {
    int idx = blockIdx.x * blockDim.x + threadIdx.x;
    const int qn = BDIM * DDIM;
    const int total = qn + KDIM * DDIM;
    if (idx >= total) return;
    if (idx < qn) {
        int b = idx >> 9, d = idx & 511;
        float v = q[(size_t)b * DP1 + 1 + d];
        __bf16 h = (__bf16)v;
        qhi[idx] = h;
        qlo[idx] = (__bf16)(v - (float)h);
    } else {
        int i = idx - qn;
        int k = i >> 9, d = i & 511;
        float v = p[(size_t)k * DP1 + 1 + d];
        __bf16 h = (__bf16)v;
        phi[i] = h;
        plo[i] = (__bf16)(v - (float)h);
    }
}

// A fragment (16x32 bf16): two 8-element runs at +0 and +16 elements.
__device__ __forceinline__ v16bf ldA(const __bf16* pp) {
    v8bf x = *(const v8bf*)pp;
    v8bf y = *(const v8bf*)(pp + 16);
    v16bf r;
    #pragma unroll
    for (int i = 0; i < 8; ++i) { r[i] = x[i]; r[i + 8] = y[i]; }
    return r;
}
// B fragment (32x16 bf16): two 8-element runs at +0 and +16 elements.
__device__ __forceinline__ v16bf ldB(const __bf16* pp) {
    v8bf x = *(const v8bf*)pp;
    v8bf y = *(const v8bf*)(pp + 16);
    v16bf r;
    #pragma unroll
    for (int i = 0; i < 8; ++i) { r[i] = x[i]; r[i + 8] = y[i]; }
    return r;
}

__device__ __forceinline__ void online_upd(float dot, float q0, float p0,
                                           float& m, float& s,
                                           float* tl, bool is_target) {
    float arg   = fmaxf(q0 * p0 - dot, 1.0f + 1e-6f);            // -C*inner, clipped
    float dist  = __logf(arg + __fsqrt_rn(arg * arg - 1.0f));    // acosh
    float logit = -10.0f * dist;                                  // -dist/TEMP
    if (is_target) *tl = logit;
    float nm = fmaxf(m, logit);
    s = s * __expf(m - nm) + __expf(logit - nm);
    m = nm;
}

__device__ __forceinline__ void merge_ms(float& m, float& s, float om, float os) {
    float nm = fmaxf(m, om);
    s = s * __expf(m - nm) + os * __expf(om - nm);
    m = nm;
}

// ---------------------------------------------------------------------------
// Phase 1: fused split-bf16 WMMA GEMM + acosh + online softmax.
// WG = 32 query rows x all 4096 columns; 8 waves; each wave owns a 32x32
// output block per 256-column tile (2x2 register tile -> 12 WMMAs per
// 8 fragment loads), 16 column tiles.
// ---------------------------------------------------------------------------
__launch_bounds__(256)
__global__ void hyp_proto_loss_kernel(const float* __restrict__ q,
                                      const float* __restrict__ p,
                                      const int*   __restrict__ tgt,
                                      const __bf16* __restrict__ qhi,
                                      const __bf16* __restrict__ qlo,
                                      const __bf16* __restrict__ phi,
                                      const __bf16* __restrict__ plo,
                                      float* __restrict__ partial) {
    __shared__ __bf16 sQh[32 * DDIM];   // Q-tile hi, row-major [32][512] (32 KB)
    __shared__ __bf16 sQl[32 * DDIM];   // Q-tile lo (32 KB)
    __shared__ float  sQ0[32];
    __shared__ int    sT[32];
    __shared__ float  sTL[32];
    __shared__ float  sM[8][32];
    __shared__ float  sS[8][32];

    const int tid  = threadIdx.x;
    const int wave = tid >> 5;
    const int lane = tid & 31;
    const int l15  = lane & 15;
    const int lh   = lane >> 4;
    const int row0 = blockIdx.x * 32;

    // Stage the 32x512 Q tile (hi & lo) into LDS as 8-byte chunks.
    {
        const uint2* gh = (const uint2*)(qhi + (size_t)row0 * DDIM);
        const uint2* gl = (const uint2*)(qlo + (size_t)row0 * DDIM);
        uint2* shh = (uint2*)sQh;
        uint2* sll = (uint2*)sQl;
        #pragma unroll
        for (int i = tid; i < (32 * DDIM) / 4; i += 256) { shh[i] = gh[i]; sll[i] = gl[i]; }
    }
    if (tid < 32) {
        sQ0[tid] = q[(size_t)(row0 + tid) * DP1];
        sT[tid]  = tgt[row0 + tid];
    }
    __syncthreads();

    // C layout: lanes 0-15 -> M = rowbase..rowbase+7 with rowbase = lh*8.
    const int rowbase = lh * 8;
    float q0r[2][8]; int tr[2][8];
    #pragma unroll
    for (int t = 0; t < 2; ++t)
        #pragma unroll
        for (int r = 0; r < 8; ++r) {
            q0r[t][r] = sQ0[t * 16 + rowbase + r];
            tr[t][r]  = sT[t * 16 + rowbase + r];
        }

    float m0[8], s0[8], m1[8], s1[8];
    #pragma unroll
    for (int r = 0; r < 8; ++r) { m0[r] = -1e30f; s0[r] = 0.f; m1[r] = -1e30f; s1[r] = 0.f; }

    // A fragment bases (constant across j).
    const __bf16* abh0 = sQh + l15 * DDIM + lh * 8;
    const __bf16* abl0 = sQl + l15 * DDIM + lh * 8;
    const __bf16* abh1 = abh0 + 16 * DDIM;
    const __bf16* abl1 = abl0 + 16 * DDIM;

    for (int j = 0; j < KDIM / 256; ++j) {
        const int n0   = j * 256 + wave * 32;
        const int colA = n0 + l15;          // column tile 0
        const int colB = n0 + 16 + l15;     // column tile 1
        v8f acc00 = {}, acc01 = {}, acc10 = {}, acc11 = {};

        const __bf16* bbh0 = phi + (size_t)colA * DDIM + lh * 8;
        const __bf16* bbl0 = plo + (size_t)colA * DDIM + lh * 8;
        const __bf16* bbh1 = phi + (size_t)colB * DDIM + lh * 8;
        const __bf16* bbl1 = plo + (size_t)colB * DDIM + lh * 8;

        #pragma unroll 2
        for (int kk = 0; kk < DDIM; kk += 32) {
            v16bf ah0 = ldA(abh0 + kk), al0 = ldA(abl0 + kk);
            v16bf ah1 = ldA(abh1 + kk), al1 = ldA(abl1 + kk);
            v16bf bh0 = ldB(bbh0 + kk), bl0 = ldB(bbl0 + kk);
            v16bf bh1 = ldB(bbh1 + kk), bl1 = ldB(bbl1 + kk);

            acc00 = __builtin_amdgcn_wmma_f32_16x16x32_bf16(false, ah0, false, bh0, (short)0, acc00, false, false);
            acc01 = __builtin_amdgcn_wmma_f32_16x16x32_bf16(false, ah0, false, bh1, (short)0, acc01, false, false);
            acc10 = __builtin_amdgcn_wmma_f32_16x16x32_bf16(false, ah1, false, bh0, (short)0, acc10, false, false);
            acc11 = __builtin_amdgcn_wmma_f32_16x16x32_bf16(false, ah1, false, bh1, (short)0, acc11, false, false);

            acc00 = __builtin_amdgcn_wmma_f32_16x16x32_bf16(false, ah0, false, bl0, (short)0, acc00, false, false);
            acc01 = __builtin_amdgcn_wmma_f32_16x16x32_bf16(false, ah0, false, bl1, (short)0, acc01, false, false);
            acc10 = __builtin_amdgcn_wmma_f32_16x16x32_bf16(false, ah1, false, bl0, (short)0, acc10, false, false);
            acc11 = __builtin_amdgcn_wmma_f32_16x16x32_bf16(false, ah1, false, bl1, (short)0, acc11, false, false);

            acc00 = __builtin_amdgcn_wmma_f32_16x16x32_bf16(false, al0, false, bh0, (short)0, acc00, false, false);
            acc01 = __builtin_amdgcn_wmma_f32_16x16x32_bf16(false, al0, false, bh1, (short)0, acc01, false, false);
            acc10 = __builtin_amdgcn_wmma_f32_16x16x32_bf16(false, al1, false, bh0, (short)0, acc10, false, false);
            acc11 = __builtin_amdgcn_wmma_f32_16x16x32_bf16(false, al1, false, bh1, (short)0, acc11, false, false);
        }

        // Fused epilogue on the 2x2 block.
        const float p0A = p[(size_t)colA * DP1];
        const float p0B = p[(size_t)colB * DP1];
        #pragma unroll
        for (int r = 0; r < 8; ++r) {
            online_upd(acc00[r], q0r[0][r], p0A, m0[r], s0[r], &sTL[rowbase + r],      colA == tr[0][r]);
            online_upd(acc01[r], q0r[0][r], p0B, m0[r], s0[r], &sTL[rowbase + r],      colB == tr[0][r]);
            online_upd(acc10[r], q0r[1][r], p0A, m1[r], s1[r], &sTL[16 + rowbase + r], colA == tr[1][r]);
            online_upd(acc11[r], q0r[1][r], p0B, m1[r], s1[r], &sTL[16 + rowbase + r], colB == tr[1][r]);
        }
    }

    // Combine (max,sum) across the 16 lanes of each half (xor stays in-half).
    #pragma unroll
    for (int r = 0; r < 8; ++r) {
        for (int off = 1; off < 16; off <<= 1) {
            merge_ms(m0[r], s0[r], __shfl_xor(m0[r], off, 32), __shfl_xor(s0[r], off, 32));
            merge_ms(m1[r], s1[r], __shfl_xor(m1[r], off, 32), __shfl_xor(s1[r], off, 32));
        }
    }
    if (l15 == 0) {
        #pragma unroll
        for (int r = 0; r < 8; ++r) {
            sM[wave][rowbase + r]      = m0[r]; sS[wave][rowbase + r]      = s0[r];
            sM[wave][16 + rowbase + r] = m1[r]; sS[wave][16 + rowbase + r] = s1[r];
        }
    }
    __syncthreads();

    // Wave 0: merge 8 waves per row (32 rows -> 32 lanes), reduce, emit.
    if (wave == 0) {
        const int row = lane;
        float M = -1e30f, S = 0.f;
        #pragma unroll
        for (int w = 0; w < 8; ++w) merge_ms(M, S, sM[w][row], sS[w][row]);
        float nll = (M + __logf(S)) - sTL[row];
        for (int off = 1; off < 32; off <<= 1) nll += __shfl_xor(nll, off, 32);
        if (lane == 0) partial[blockIdx.x] = nll;   // deterministic: no atomics
    }
}

// ---------------------------------------------------------------------------
// Phase 2: fixed-order tree reduction of the 256 per-block sums -> mean NLL.
// ---------------------------------------------------------------------------
__global__ void reduce_kernel(const float* __restrict__ partial, float* __restrict__ out) {
    __shared__ float sm[256];
    int tid = threadIdx.x;
    sm[tid] = partial[tid];
    __syncthreads();
    for (int off = 128; off > 0; off >>= 1) {
        if (tid < off) sm[tid] += sm[tid + off];
        __syncthreads();
    }
    if (tid == 0) out[0] = sm[0] * (1.0f / (float)BDIM);
}

extern "C" void kernel_launch(void* const* d_in, const int* in_sizes, int n_in,
                              void* d_out, int out_size, void* d_ws, size_t ws_size,
                              hipStream_t stream) {
    (void)in_sizes; (void)n_in; (void)out_size; (void)ws_size;
    const float* q = (const float*)d_in[0];   // [B, 513] f32
    const float* p = (const float*)d_in[1];   // [K, 513] f32
    const int*   t = (const int*)d_in[2];     // [B] i32

    // Workspace layout (~24 MB + 1 KB):
    __bf16* qhi = (__bf16*)d_ws;
    __bf16* qlo = qhi + (size_t)BDIM * DDIM;
    __bf16* phi = qlo + (size_t)BDIM * DDIM;
    __bf16* plo = phi + (size_t)KDIM * DDIM;
    float* partial = (float*)(plo + (size_t)KDIM * DDIM);

    const int total = (BDIM + KDIM) * DDIM;
    split_bf16_kernel<<<(total + 255) / 256, 256, 0, stream>>>(q, p, qhi, qlo, phi, plo);
    hyp_proto_loss_kernel<<<BDIM / 32, 256, 0, stream>>>(q, p, t, qhi, qlo, phi, plo, partial);
    reduce_kernel<<<1, 256, 0, stream>>>(partial, (float*)d_out);
}
